// AttentionBlock_48421461295245
// MI455X (gfx1250) — matrix operand order repaired and checked by reading, not compiled, not run on hardware
//
#include <hip/hip_runtime.h>

typedef __attribute__((ext_vector_type(16))) __bf16       v16bf;
typedef __attribute__((ext_vector_type(8)))  float        v8f;
typedef __attribute__((ext_vector_type(8)))  unsigned int v8u;

#define C_ 64
#define L_ 512
#define D_ 128
#define INVSQRTD 0.08838834764831845f  // 1/sqrt(128)
#define KW 584   // padded k-stride for weight LDS tile (576 + 8)
#define KB 296   // padded k-stride for im2col half tile (288 + 8)
#define ROWS 4   // output rows per conv block (amortizes weight staging)

// ---------------- WMMA helpers (CDNA5 16x16x32 bf16, f32 accum) ----------------

__device__ __forceinline__ v8f wmma_bf16(v16bf a, v16bf b, v8f c) {
  return __builtin_amdgcn_wmma_f32_16x16x32_bf16(false, a, false, b, (short)0, c,
                                                 false, false);
}

// A-matrix 16x32 bf16 fragment from LDS, source row-major [m][k], stride in elems.
__device__ __forceinline__ v16bf ldfragA(const __bf16* p, int stride, int lane) {
  const int m = lane & 15, hi = lane >> 4;
  const __bf16* row = p + m * stride;
  v8u u;
#pragma unroll
  for (int v = 0; v < 8; ++v) {
    const int k = ((v >> 2) << 4) + (hi << 3) + ((v & 3) << 1);
    u[v] = *(const unsigned int*)(row + k);
  }
  return __builtin_bit_cast(v16bf, u);
}

// B-matrix 32x16 bf16 fragment; source is B^T stored row-major [n][k].
__device__ __forceinline__ v16bf ldfragB(const __bf16* p, int stride, int lane) {
  const int n = lane & 15, hi = lane >> 4;
  const __bf16* row = p + n * stride;
  v8u u;
#pragma unroll
  for (int v = 0; v < 8; ++v) {
    const int k = ((v >> 2) << 4) + (hi << 3) + ((v & 3) << 1);
    u[v] = *(const unsigned int*)(row + k);
  }
  return __builtin_bit_cast(v16bf, u);
}

__device__ __forceinline__ float half16_max(float x) {
#pragma unroll
  for (int off = 1; off < 16; off <<= 1) x = fmaxf(x, __shfl_xor(x, off, 32));
  return x;
}
__device__ __forceinline__ float half16_sum(float x) {
#pragma unroll
  for (int off = 1; off < 16; off <<= 1) x += __shfl_xor(x, off, 32);
  return x;
}

// ---------------- Stage 0: convert conv weights f32 -> bf16 (once) ----------------
__global__ __launch_bounds__(256) void wcvt_kernel(const float* __restrict__ wq,
                                                   const float* __restrict__ wk,
                                                   const float* __restrict__ wv,
                                                   __bf16* __restrict__ dst) {
  const int i = blockIdx.x * 256 + threadIdx.x;
  const int W = C_ * C_ * 9;  // 36864
  if (i < W)
    dst[i] = (__bf16)wq[i];
  else if (i < 2 * W)
    dst[i] = (__bf16)wk[i - W];
  else if (i < 3 * W)
    dst[i] = (__bf16)wv[i - 2 * W];
}

// ---------------- Stage 1: 3x3 SAME conv as implicit-GEMM WMMA bf16 ----------------
// grid = (L/ROWS, B, 3 tensors). Block: GEMM  M=64 c_out x N=128 d x K=576 (c_in*9)
// for ROWS consecutive output rows. Weights staged to LDS once per block; im2col
// B^T rebuilt per (row, K-half of 288) in a reused 75.8 KB buffer.
__global__ __launch_bounds__(256) void conv3x3_wmma_kernel(
    const float* __restrict__ xq, const float* __restrict__ xk,
    const float* __restrict__ xv, const __bf16* __restrict__ wall,
    __bf16* __restrict__ oq, __bf16* __restrict__ ok, __bf16* __restrict__ ov) {
  __shared__ alignas(16) __bf16 wlds[C_ * KW];   // 74.75 KB
  __shared__ alignas(16) __bf16 blds[D_ * KB];   // 75.8 KB
  const int z = blockIdx.z;
  const float* x = (z == 0) ? xq : (z == 1) ? xk : xv;
  __bf16* out = (z == 0) ? oq : (z == 1) ? ok : ov;
  const __bf16* wb = wall + (size_t)z * (C_ * C_ * 9);

  const int lbase = blockIdx.x * ROWS;
  const int b = blockIdx.y;
  const int tid = threadIdx.x;
  const int lane = tid & 31;
  const int wv = tid >> 5;

  // Stage all weights (bf16 copy, dword-granular, coalesced).
  for (int i = tid; i < C_ * 288; i += 256) {  // 288 dword-pairs per row
    const int co = i / 288;
    const int kp = (i % 288) << 1;
    *(unsigned int*)(wlds + co * KW + kp) =
        *(const unsigned int*)(wb + co * 576 + kp);
  }

  const int mt = wv >> 1;        // M tile 0..3 (16 c_out each)
  const int nt0 = (wv & 1) * 4;  // first of 4 N tiles (16 d each)
  const v8f vzero = {};
  const int hi8 = (lane >> 4) << 3;
  const int nn = lane & 15;

  for (int rr = 0; rr < ROWS; ++rr) {
    const int l0 = lbase + rr;
    v8f acc[4];
#pragma unroll
    for (int j = 0; j < 4; ++j) acc[j] = vzero;

    for (int h = 0; h < 2; ++h) {
      __syncthreads();  // wlds ready (first pass); blds safe to overwrite after
      // Build im2col B^T for ci in [h*32, h*32+32):
      //   blds[d'][cil*9 + dl*3 + dw] = x[b, ci, l0+dl-1, d'+dw-1]
      for (int i = tid; i < 32 * 3 * D_; i += 256) {
        const int d = i & 127;
        const int t2 = i >> 7;
        const int dl = t2 % 3;
        const int cil = t2 / 3;
        const int l = l0 - 1 + dl;
        float v = 0.0f;
        if ((unsigned)l < (unsigned)L_)
          v = x[(((size_t)b * C_ + (h * 32 + cil)) * L_ + l) * D_ + d];
        const __bf16 bv = (__bf16)v;
        const int kbase = cil * 9 + dl * 3;
        if (d + 1 < D_) blds[(d + 1) * KB + kbase + 0] = bv;  // dw=0: d' = d+1
        blds[d * KB + kbase + 1] = bv;                        // dw=1: d' = d
        if (d - 1 >= 0) blds[(d - 1) * KB + kbase + 2] = bv;  // dw=2: d' = d-1
        if (d == 0) blds[0 * KB + kbase + 0] = (__bf16)0.0f;      // x[-1]
        if (d == 127) blds[127 * KB + kbase + 2] = (__bf16)0.0f;  // x[128]
      }
      __syncthreads();

#pragma unroll
      for (int kc = 0; kc < 9; ++kc) {
        const v16bf af =
            ldfragA(wlds + mt * 16 * KW + h * 288 + kc * 32, KW, lane);
#pragma unroll
        for (int j = 0; j < 4; ++j) {
          const v16bf bf =
              ldfragB(blds + (nt0 + j) * 16 * KB + kc * 32, KB, lane);
          acc[j] = wmma_bf16(af, bf, acc[j]);
        }
      }
    }

    // Store this row: C layout lane -> (n = d col, vgpr r -> c_out row).
#pragma unroll
    for (int j = 0; j < 4; ++j) {
      const int d = (nt0 + j) * 16 + nn;
#pragma unroll
      for (int r = 0; r < 8; ++r) {
        const int co = mt * 16 + r + hi8;
        out[(((size_t)b * C_ + co) * L_ + l0) * D_ + d] = (__bf16)acc[j][r];
      }
    }
  }
}

// ---------------- Stage 2: per-(b,c) flash attention with WMMA bf16 ----------------
__global__ __launch_bounds__(256) void flash_attn_kernel(
    const __bf16* __restrict__ qb, const __bf16* __restrict__ kb,
    const __bf16* __restrict__ vb, float* __restrict__ ob) {
  __shared__ alignas(16) __bf16 klds[128 * 130];
  __shared__ alignas(16) __bf16 vtlds[128 * 130];
  __shared__ alignas(16) __bf16 plds[128 * 130];

  const int bc = blockIdx.x;
  const int r0 = blockIdx.y * 128;
  const int tid = threadIdx.x;
  const int lane = tid & 31;
  const int wv = tid >> 5;
  const size_t base = (size_t)bc * L_ * D_;

  for (int i = tid; i < 128 * 64; i += 256) {
    const int row = i >> 6;
    const int cp = (i & 63) << 1;
    *(unsigned int*)(plds + row * 130 + cp) =
        *(const unsigned int*)(qb + base + (size_t)(r0 + row) * D_ + cp);
  }
  __syncthreads();
  v16bf qfrag[4];
#pragma unroll
  for (int kc = 0; kc < 4; ++kc)
    qfrag[kc] = ldfragA(plds + wv * 16 * 130 + kc * 32, 130, lane);
  __syncthreads();

  const v8f vzero = {};
  v8f oacc[8];
#pragma unroll
  for (int j = 0; j < 8; ++j) oacc[j] = vzero;
  float rm[8], rl[8];
#pragma unroll
  for (int r = 0; r < 8; ++r) { rm[r] = -1.0e30f; rl[r] = 0.0f; }

  for (int t = 0; t < 4; ++t) {  // 4 KV tiles of 128
    const size_t kvbase = base + (size_t)t * 128 * D_;
    for (int i = tid; i < 128 * 64; i += 256) {
      const int row = i >> 6;
      const int cp = (i & 63) << 1;
      const unsigned int uk =
          *(const unsigned int*)(kb + kvbase + (size_t)row * D_ + cp);
      *(unsigned int*)(klds + row * 130 + cp) = uk;
      const unsigned int uv =
          *(const unsigned int*)(vb + kvbase + (size_t)row * D_ + cp);
      vtlds[cp * 130 + row] =
          __builtin_bit_cast(__bf16, (unsigned short)(uv & 0xffffu));
      vtlds[(cp + 1) * 130 + row] =
          __builtin_bit_cast(__bf16, (unsigned short)(uv >> 16));
    }
    __syncthreads();

    v8f sacc[8];
#pragma unroll
    for (int j = 0; j < 8; ++j) sacc[j] = vzero;
#pragma unroll
    for (int kc = 0; kc < 4; ++kc) {
#pragma unroll
      for (int j = 0; j < 8; ++j) {
        const v16bf bk = ldfragB(klds + j * 16 * 130 + kc * 32, 130, lane);
        sacc[j] = wmma_bf16(qfrag[kc], bk, sacc[j]);
      }
    }

#pragma unroll
    for (int r = 0; r < 8; ++r) {
      float mx = sacc[0][r];
#pragma unroll
      for (int j = 1; j < 8; ++j) mx = fmaxf(mx, sacc[j][r]);
      mx = half16_max(mx);
      const float newm = fmaxf(rm[r], mx);
      const float corr = __expf((rm[r] - newm) * INVSQRTD);
      float ls = 0.0f;
#pragma unroll
      for (int j = 0; j < 8; ++j) {
        const float p = __expf((sacc[j][r] - newm) * INVSQRTD);
        sacc[j][r] = p;
        ls += p;
      }
      ls = half16_sum(ls);
      rl[r] = rl[r] * corr + ls;
      rm[r] = newm;
#pragma unroll
      for (int j = 0; j < 8; ++j) oacc[j][r] *= corr;
    }

    {
      const int hi8 = (lane >> 4) << 3;
      const int nn = lane & 15;
#pragma unroll
      for (int r = 0; r < 8; ++r) {
        const int row = wv * 16 + r + hi8;
#pragma unroll
        for (int j = 0; j < 8; ++j)
          plds[row * 130 + j * 16 + nn] = (__bf16)sacc[j][r];
      }
    }

#pragma unroll
    for (int kc = 0; kc < 4; ++kc) {
      const v16bf pa = ldfragA(plds + wv * 16 * 130 + kc * 32, 130, lane);
#pragma unroll
      for (int j = 0; j < 8; ++j) {
        const v16bf bv = ldfragB(vtlds + j * 16 * 130 + kc * 32, 130, lane);
        oacc[j] = wmma_bf16(pa, bv, oacc[j]);
      }
    }
    __syncthreads();
  }

  const int hi8 = (lane >> 4) << 3;
  const int nn = lane & 15;
#pragma unroll
  for (int r = 0; r < 8; ++r) {
    const float inv = 1.0f / rl[r];
    const size_t row = (size_t)r0 + wv * 16 + r + hi8;
#pragma unroll
    for (int j = 0; j < 8; ++j)
      ob[base + row * D_ + j * 16 + nn] = oacc[j][r] * inv;
  }
}

// ---------------- Stage 3: 1x1 conv (channel mix) + bias + residual, fp32 ----------
__global__ __launch_bounds__(256) void outproj_kernel(
    const float* __restrict__ o, const float* __restrict__ wo,
    const float* __restrict__ bo, const float* __restrict__ query,
    float* __restrict__ out) {
  __shared__ float os[C_][128];
  const int b = blockIdx.y;
  const int pos0 = blockIdx.x * 128;
  const int tid = threadIdx.x;

  for (int i = tid; i < C_ * 128; i += 256) {
    const int ci = i >> 7;
    const int p = i & 127;
    os[ci][p] = o[(size_t)(b * C_ + ci) * (L_ * D_) + pos0 + p];
  }
  __syncthreads();

  const int p = tid & 127;
  const int cg = (tid >> 7) * 32;
  float acc[32];
#pragma unroll
  for (int c = 0; c < 32; ++c) acc[c] = bo[cg + c];

  for (int ci = 0; ci < C_; ++ci) {
    const float xv = os[ci][p];
#pragma unroll
    for (int c = 0; c < 32; ++c)
      acc[c] = fmaf(wo[(size_t)(cg + c) * C_ + ci], xv, acc[c]);
  }
#pragma unroll
  for (int c = 0; c < 32; ++c) {
    const size_t idx = (size_t)(b * C_ + cg + c) * (L_ * D_) + pos0 + p;
    out[idx] = acc[c] + query[idx];
  }
}

// ---------------- Launch ----------------

extern "C" void kernel_launch(void* const* d_in, const int* in_sizes, int n_in,
                              void* d_out, int out_size, void* d_ws, size_t ws_size,
                              hipStream_t stream) {
  (void)in_sizes; (void)n_in; (void)out_size; (void)ws_size;
  const float* query = (const float*)d_in[0];
  const float* key   = (const float*)d_in[1];
  const float* value = (const float*)d_in[2];
  const float* Wq    = (const float*)d_in[3];
  const float* Wk    = (const float*)d_in[4];
  const float* Wv    = (const float*)d_in[5];
  const float* Wo    = (const float*)d_in[6];
  const float* bo    = (const float*)d_in[7];
  float* out = (float*)d_out;

  const size_t N = (size_t)4 * C_ * L_ * D_;  // 16,777,216 elements
  const size_t W = (size_t)C_ * C_ * 9;       // 36,864 weights per conv
  __bf16* qb = (__bf16*)d_ws;                 // N bf16
  __bf16* kb = qb + N;                        // N bf16
  __bf16* vb = kb + N;                        // N bf16
  float* ob = (float*)(vb + N);               // N f32
  __bf16* wbf = (__bf16*)(ob + N);            // 3*W bf16  (total ~168 MB)

  wcvt_kernel<<<(3 * (int)W + 255) / 256, 256, 0, stream>>>(Wq, Wk, Wv, wbf);

  dim3 cgrid(L_ / ROWS, 4, 3);
  conv3x3_wmma_kernel<<<cgrid, 256, 0, stream>>>(query, key, value, wbf, qb, kb,
                                                 vb);

  dim3 agrid(4 * C_, L_ / 128);
  flash_attn_kernel<<<agrid, 256, 0, stream>>>(qb, kb, vb, ob);

  dim3 pgrid((L_ * D_) / 128, 4);
  outproj_kernel<<<pgrid, 256, 0, stream>>>(ob, Wo, bo, query, out);
}
